// CrossTransformerBlock_31310311588316
// MI455X (gfx1250) — matrix operand order repaired and checked by reading, not compiled, not run on hardware
//
#include <hip/hip_runtime.h>
#include <hip/hip_bf16.h>

// ---------------------------------------------------------------------------
// Problem constants (from reference)
// ---------------------------------------------------------------------------
constexpr int BB   = 2;
constexpr int NQ   = 4096;
constexpr int NN   = 8192;
constexpr int DIM  = 256;   // == DIN
constexpr int KNN  = 16;
constexpr int KK2  = 8;
constexpr int NB   = KNN + KK2 + 1;   // 25 neighbor rows per query
constexpr int QT   = 2;               // queries per fused block
constexpr int MROWS = 32 * QT;        // padded rows per fused block
constexpr int LDS_STRIDE = 264;       // halves; 528B row -> 16B aligned, bank-spread

typedef _Float16 v4h  __attribute__((ext_vector_type(4)));
typedef _Float16 v8h  __attribute__((ext_vector_type(8)));
typedef _Float16 v16h __attribute__((ext_vector_type(16)));
typedef float    v8f  __attribute__((ext_vector_type(8)));

// ---------------------------------------------------------------------------
// Tensor Data Mover (gather mode) availability / arity probing
// ---------------------------------------------------------------------------
#if defined(__has_builtin)
#  if __has_builtin(__builtin_amdgcn_tensor_load_to_lds)
#    define HAVE_TDM 1
#  endif
#endif
#ifndef HAVE_TDM
#  define HAVE_TDM 0
#endif
#if HAVE_TDM
#  if __has_include(<hip/amd_detail/amd_gfx1250_TDM.h>)
#    define TDM_SIX_ARGS 1
#  else
#    define TDM_SIX_ARGS 0
#  endif
typedef unsigned int u32x4 __attribute__((ext_vector_type(4)));
typedef int          i32x4 __attribute__((ext_vector_type(4)));
typedef int          i32x8 __attribute__((ext_vector_type(8)));

// Gather 16 rows (16-bit indices) of 256 x 2-byte elements from a row-major
// [NN][256] f16 tensor into LDS, with pad_interval=128 DWORDs / pad_amount=4
// DWORDs so the LDS row stride is 512+16 = 528 bytes (== LDS_STRIDE halves).
__device__ __forceinline__ void tdm_gather16_rows(unsigned lds_byte_off,
                                                  const _Float16* gbase,
                                                  const int* idx16 /*16 ints*/) {
    unsigned long long ga = (unsigned long long)(size_t)gbase;
    u32x4 g0;
    g0.x = 1u | (1u << 31);                                   // count=1, gather_mode, 16b idx
    g0.y = lds_byte_off;                                      // D#.lds_addr
    g0.z = (unsigned)ga;                                      // global_addr[31:0]
    g0.w = (unsigned)((ga >> 32) & 0x01ffffffu) | (2u << 30); // global_addr[56:32], type=2
    i32x8 g1;
    g1[0] = (1 << 16) | (1 << 20) | (6 << 22) | (3 << 25);    // 2B elems; pad 128dw by 4dw
    g1[1] = (int)(256u << 16);                                // tensor_dim0 = 256
    g1[2] = (int)((unsigned)NN << 16);                        // tensor_dim1 = 8192
    g1[3] = (int)(256u << 16);                                // tile_dim0 = 256
    g1[4] = 16;                                               // tile_dim1 = 16 indices
    g1[5] = 256;                                              // tensor_dim0_stride = 256
    g1[6] = 0;
    g1[7] = 0;
    i32x4 g2, g3;
#pragma unroll
    for (int j = 0; j < 4; ++j) {
        g2[j] = __builtin_amdgcn_readfirstlane(idx16[2 * j]     | (idx16[2 * j + 1] << 16));
        g3[j] = __builtin_amdgcn_readfirstlane(idx16[8 + 2 * j] | (idx16[9 + 2 * j] << 16));
    }
#if TDM_SIX_ARGS
    i32x8 gz = {0, 0, 0, 0, 0, 0, 0, 0};
    __builtin_amdgcn_tensor_load_to_lds(g0, g1, g2, g3, gz, 0);
#else
    __builtin_amdgcn_tensor_load_to_lds(g0, g1, g2, g3, 0);
#endif
}
#endif  // HAVE_TDM

// ---------------------------------------------------------------------------
// WMMA fragment helpers (CDNA5 16x16x32 f16, f32 accumulate)
// ---------------------------------------------------------------------------
__device__ __forceinline__ v16h frag_a_lds(const _Float16* base, int row0, int k0, int lane) {
    const int row = row0 + (lane & 15);
    const int hi  = lane >> 4;
    const _Float16* p = base + row * LDS_STRIDE + k0 + hi * 8;
    v8h lo = *(const v8h*)p;
    v8h hh = *(const v8h*)(p + 16);
    return __builtin_shufflevector(lo, hh, 0,1,2,3,4,5,6,7,8,9,10,11,12,13,14,15);
}

__device__ __forceinline__ v16h frag_b_wt(const _Float16* wt /*[out][in] f16*/, int n0, int k0, int lane) {
    const int n    = n0 + (lane & 15);
    const int koff = k0 + (lane >> 4) * 8;
    const _Float16* p = wt + n * DIM + koff;
    v8h lo = *(const v8h*)p;
    v8h hh = *(const v8h*)(p + 16);
    return __builtin_shufflevector(lo, hh, 0,1,2,3,4,5,6,7,8,9,10,11,12,13,14,15);
}

__device__ __forceinline__ void preload_b(const _Float16* wt, int n0, int lane, v16h bf[8]) {
#pragma unroll
    for (int kt = 0; kt < 8; ++kt) bf[kt] = frag_b_wt(wt, n0, kt * 32, lane);
}

__device__ __forceinline__ v8f gemm_tile_b(const _Float16* aBase, int mRow0,
                                           const v16h bf[8], int lane) {
    v8f acc = {};
#pragma unroll
    for (int kt = 0; kt < 8; ++kt) {
        v16h a = frag_a_lds(aBase, mRow0, kt * 32, lane);
        acc = __builtin_amdgcn_wmma_f32_16x16x32_f16(
            false, a, false, bf[kt], (short)0, acc, false, false);
    }
    return acc;
}

// ---------------------------------------------------------------------------
// K0: convert + transpose a 256x256 f32 weight [in][out] -> f16 [out][in]
// ---------------------------------------------------------------------------
__global__ __launch_bounds__(256) void prep_weight_kernel(const float* __restrict__ src,
                                                          _Float16* __restrict__ dst) {
    int idx = blockIdx.x * 256 + threadIdx.x;
    if (idx < DIM * DIM) {
        int i = idx >> 8;       // in
        int o = idx & 255;      // out
        dst[o * DIM + i] = (_Float16)src[idx];
    }
}

// ---------------------------------------------------------------------------
// K1: global-token projections  q/kg/vg = lat_rep @ {w_qs,w_kg,w_vg}  (tiny)
// ---------------------------------------------------------------------------
__global__ __launch_bounds__(256) void globals_kernel(
    const float* __restrict__ lat, const float* __restrict__ wq,
    const float* __restrict__ wk, const float* __restrict__ wv,
    float* __restrict__ qf, float* __restrict__ kgf, float* __restrict__ vgf) {
    int n = threadIdx.x;
    for (int b = 0; b < BB; ++b) {
        float aq = 0.f, ak = 0.f, av = 0.f;
        for (int k = 0; k < DIM; ++k) {
            float l = lat[b * DIM + k];
            aq += l * wq[k * DIM + n];
            ak += l * wk[k * DIM + n];
            av += l * wv[k * DIM + n];
        }
        qf[b * DIM + n] = aq;
        kgf[b * DIM + n] = ak;
        vgf[b * DIM + n] = av;
    }
}

// ---------------------------------------------------------------------------
// K2: brute-force KNN; distances in VGPRs, wave shfl_xor argmin, 2 barriers/rnd
// ---------------------------------------------------------------------------
__global__ __launch_bounds__(256) void knn_kernel(const float* __restrict__ xyz_q,
                                                  const float* __restrict__ xyz,
                                                  int* __restrict__ knn) {
    __shared__ float rv[8];
    __shared__ int   ri[8];
    __shared__ float q3[3];
    __shared__ int   sWin;

    const int b = blockIdx.x / NQ;
    const int q = blockIdx.x % NQ;
    const int tid = threadIdx.x;
    const int lane = tid & 31, wave = tid >> 5;

    if (tid < 3) q3[tid] = xyz_q[(b * NQ + q) * 3 + tid];
    __syncthreads();
    const float qx = q3[0], qy = q3[1], qz = q3[2];

    float d[32];
#pragma unroll
    for (int j = 0; j < 32; ++j) {
        int i = j * 256 + tid;
        const float* p = xyz + ((size_t)b * NN + i) * 3;
        float dx = qx - p[0], dy = qy - p[1], dz = qz - p[2];
        d[j] = dx * dx + dy * dy + dz * dz;
    }

    for (int t = 0; t < KNN; ++t) {
        float best = 3.4e38f; int bj = 0;
#pragma unroll
        for (int j = 0; j < 32; ++j)
            if (d[j] < best) { best = d[j]; bj = j; }
        int bi = bj * 256 + tid;
#pragma unroll
        for (int off = 16; off > 0; off >>= 1) {
            float ov = __shfl_xor(best, off, 32);
            int   oi = __shfl_xor(bi, off, 32);
            if (ov < best || (ov == best && oi < bi)) { best = ov; bi = oi; }
        }
        if (lane == 0) { rv[wave] = best; ri[wave] = bi; }
        __syncthreads();
        if (tid == 0) {
            float bb = rv[0]; int bbi = ri[0];
            for (int w = 1; w < 8; ++w)
                if (rv[w] < bb || (rv[w] == bb && ri[w] < bbi)) { bb = rv[w]; bbi = ri[w]; }
            knn[((size_t)b * NQ + q) * KNN + t] = bbi;
            sWin = bbi;
        }
        __syncthreads();
        int win = sWin;
        if ((win & 255) == tid) d[win >> 8] = 3.4e38f;   // owner invalidates
    }
}

// ---------------------------------------------------------------------------
// K3: dual projection GEMM: o1 = A@w1, o2 = A@w2 (f16 out). Block = 64 rows
// (rows is an exact multiple of 64 -> no tail guards, no saveexec chains).
// ---------------------------------------------------------------------------
__global__ __launch_bounds__(256) void dual_proj_kernel(const float* __restrict__ A,
                                                        const _Float16* __restrict__ wt1,
                                                        const _Float16* __restrict__ wt2,
                                                        _Float16* __restrict__ o1,
                                                        _Float16* __restrict__ o2) {
    __shared__ _Float16 sA[64 * LDS_STRIDE];
    const int m0  = blockIdx.x * 64;
    const int tid = threadIdx.x;

    for (int e = tid; e < 64 * 64; e += 256) {       // float4 chunks
        int r = e >> 6, cc = (e & 63) * 4;
        const float4 f = *(const float4*)(A + (size_t)(m0 + r) * DIM + cc);
        v4h hv = {(_Float16)f.x, (_Float16)f.y, (_Float16)f.z, (_Float16)f.w};
        *(v4h*)(void*)(sA + r * LDS_STRIDE + cc) = hv;
    }
    __syncthreads();

    const int lane = tid & 31, wave = tid >> 5, hi = lane >> 4;
    for (int ntl = 0; ntl < 2; ++ntl) {
        const int nt = wave * 2 + ntl;
        const int n0 = nt * 16;
        const int nn = n0 + (lane & 15);
        for (int mat = 0; mat < 2; ++mat) {
            const _Float16* wt = mat ? wt2 : wt1;
            _Float16* o = mat ? o2 : o1;
            v16h bf[8];
            preload_b(wt, n0, lane, bf);
            for (int mt = 0; mt < 4; ++mt) {
                v8f acc = gemm_tile_b(sA, mt * 16, bf, lane);
#pragma unroll
                for (int vv = 0; vv < 8; ++vv) {
                    int m = m0 + mt * 16 + vv + hi * 8;
                    o[(size_t)m * DIM + nn] = (_Float16)acc[vv];
                }
            }
        }
    }
}

// ---------------------------------------------------------------------------
// K4: fused attention, 2 queries per block (M = 64 padded rows).
// sK: k -> x -> logits ; sV: v -> v+pos ; sH: grid-A / hidden (aliased).
// k/v gather done by the Tensor Data Mover (gather mode) when available.
// ---------------------------------------------------------------------------
__global__ __launch_bounds__(256) void fused_attn_kernel(
    const int* __restrict__ knn,
    const _Float16* __restrict__ pk16, const _Float16* __restrict__ pv16,
    const float* __restrict__ grid_feat, const float* __restrict__ closest,
    const float* __restrict__ xyz, const float* __restrict__ xyz_q,
    const float* __restrict__ qf, const float* __restrict__ kgf,
    const float* __restrict__ vgf,
    const _Float16* __restrict__ wt_kc, const _Float16* __restrict__ wt_vc,
    const float* __restrict__ dw1, const float* __restrict__ db1,
    const _Float16* __restrict__ wt_d2, const float* __restrict__ db2,
    const _Float16* __restrict__ wt_g1, const float* __restrict__ gb1,
    const _Float16* __restrict__ wt_g2, const float* __restrict__ gb2,
    float* __restrict__ out) {
    __shared__ _Float16 sK[MROWS * LDS_STRIDE];
    __shared__ _Float16 sV[MROWS * LDS_STRIDE];
    __shared__ _Float16 sH[MROWS * LDS_STRIDE];   // doubles as grid-feat A tile
    __shared__ float    sD[MROWS * 4];
    __shared__ int      sIdx[QT * KNN];

    const int gq0 = blockIdx.x * QT;        // global query index (b folded in)
    const int b   = gq0 >> 12;              // / NQ
    const int tid = threadIdx.x;
    const int lane = tid & 31, wave = tid >> 5, hi = lane >> 4;

    if (tid < QT * KNN) {
        int qi = tid >> 4, r = tid & 15;
        sIdx[tid] = knn[(size_t)(gq0 + qi) * KNN + r];
    }
    __syncthreads();

#if HAVE_TDM
    // --- TDM gather-mode DMA: 16 indexed rows per (query, tensor) ---
    if (wave == 0) {
        const _Float16* kbase = pk16 + (size_t)b * NN * DIM;
        const _Float16* vbase = pv16 + (size_t)b * NN * DIM;
        for (int qi = 0; qi < QT; ++qi) {
            unsigned kOff = (unsigned)(size_t)(void*)(sK + (qi * 32) * LDS_STRIDE);
            unsigned vOff = (unsigned)(size_t)(void*)(sV + (qi * 32) * LDS_STRIDE);
            tdm_gather16_rows(kOff, kbase, sIdx + qi * KNN);
            tdm_gather16_rows(vOff, vbase, sIdx + qi * KNN);
        }
        __builtin_amdgcn_s_wait_tensorcnt(0);
    }
#else
    // --- manual gather fallback: uint4 chunks ---
    for (int e = tid; e < (QT * KNN) * 32; e += 256) {
        int r = e >> 5, c = (e & 31) * 8;
        int qi = r >> 4, rl = r & 15;
        int src = sIdx[qi * 16 + rl];
        const uint4 kv = *(const uint4*)(const void*)(pk16 + ((size_t)b * NN + src) * DIM + c);
        const uint4 vv = *(const uint4*)(const void*)(pv16 + ((size_t)b * NN + src) * DIM + c);
        int drow = qi * 32 + rl;
        *(uint4*)(void*)(sK + drow * LDS_STRIDE + c) = kv;
        *(uint4*)(void*)(sV + drow * LDS_STRIDE + c) = vv;
    }
#endif
    // --- stage grid feats into sG(=sH): 16 rows per query, rows 8..15 zero ---
    _Float16* sG = sH;
    for (int e = tid; e < (QT * 16) * 64; e += 256) {
        int r = e >> 6, cc = (e & 63) * 4;
        int qi = r >> 4, rl = r & 15;
        v4h hv = {};
        if (rl < KK2) {
            const float4 f = *(const float4*)(grid_feat + ((size_t)(gq0 + qi) * KK2 + rl) * DIM + cc);
            hv = (v4h){(_Float16)f.x, (_Float16)f.y, (_Float16)f.z, (_Float16)f.w};
        }
        *(v4h*)(void*)(sG + r * LDS_STRIDE + cc) = hv;
    }
    // --- relative position deltas ---
    if (tid < MROWS) {
        int qi = tid >> 5, ml = tid & 31;
        int bq = gq0 + qi;
        float dx = 0.f, dy = 0.f, dz = 0.f;
        if (ml < KNN + KK2) {
            const float* nb = (ml < KNN) ? (xyz + ((size_t)b * NN + sIdx[qi * 16 + ml]) * 3)
                                         : (closest + ((size_t)bq * KK2 + (ml - KNN)) * 3);
            dx = xyz_q[bq * 3 + 0] - nb[0];
            dy = xyz_q[bq * 3 + 1] - nb[1];
            dz = xyz_q[bq * 3 + 2] - nb[2];
        }
        sD[tid * 4 + 0] = dx; sD[tid * 4 + 1] = dy;
        sD[tid * 4 + 2] = dz; sD[tid * 4 + 3] = 0.f;
    }
    __syncthreads();

    // --- grid projection (WMMA) -> sK/sV rows qi*32 + 16..31 ---
    for (int ntl = 0; ntl < 2; ++ntl) {
        const int nt = wave * 2 + ntl, n0 = nt * 16;
        const int nn = n0 + (lane & 15);
        for (int mat = 0; mat < 2; ++mat) {
            const _Float16* wt = mat ? wt_vc : wt_kc;
            _Float16* dst = mat ? sV : sK;
            v16h bf[8];
            preload_b(wt, n0, lane, bf);
            for (int qi = 0; qi < QT; ++qi) {
                v8f acc = gemm_tile_b(sG, qi * 16, bf, lane);
#pragma unroll
                for (int vv = 0; vv < 8; ++vv) {
                    int m = qi * 32 + 16 + vv + hi * 8;    // rows 24..31 come out 0
                    dst[m * LDS_STRIDE + nn] = (_Float16)acc[vv];
                }
            }
        }
    }
    __syncthreads();

    // --- global token row qi*32 + 24 ---
    for (int qi = 0; qi < QT; ++qi) {
        sK[(qi * 32 + 24) * LDS_STRIDE + tid] = (_Float16)kgf[b * DIM + tid];
        sV[(qi * 32 + 24) * LDS_STRIDE + tid] = (_Float16)vgf[b * DIM + tid];
    }
    __syncthreads();

    // --- pos MLP layer 1 (K=3, VALU): sH = relu(d @ dw1 + db1) ---
    {
        const int n = tid;
        float w0 = dw1[0 * DIM + n], w1 = dw1[1 * DIM + n], w2 = dw1[2 * DIM + n];
        float bb = db1[n];
#pragma unroll 4
        for (int m = 0; m < MROWS; ++m) {
            float h = sD[m * 4 + 0] * w0 + sD[m * 4 + 1] * w1 + sD[m * 4 + 2] * w2 + bb;
            sH[m * LDS_STRIDE + n] = (_Float16)fmaxf(h, 0.f);
        }
    }
    __syncthreads();

    // --- pos MLP layer 2 (WMMA), folded: sK <- q - k + pos ; sV <- v + pos ---
    for (int ntl = 0; ntl < 2; ++ntl) {
        const int nt = wave * 2 + ntl, n0 = nt * 16;
        const int nn = n0 + (lane & 15);
        v16h bf[8];
        preload_b(wt_d2, n0, lane, bf);
        const float qv = qf[b * DIM + nn];
        const float bb = db2[nn];
        for (int mt = 0; mt < MROWS / 16; ++mt) {
            v8f acc = gemm_tile_b(sH, mt * 16, bf, lane);
#pragma unroll
            for (int vv = 0; vv < 8; ++vv) {
                int m = mt * 16 + vv + hi * 8;
                int ml = m & 31;
                float pos = (ml < KNN + KK2) ? (acc[vv] + bb) : 0.f;
                float kk = (float)sK[m * LDS_STRIDE + nn];
                sK[m * LDS_STRIDE + nn] = (_Float16)(qv - kk + pos);
                sV[m * LDS_STRIDE + nn] = (_Float16)((float)sV[m * LDS_STRIDE + nn] + pos);
            }
        }
    }
    __syncthreads();

    // --- gamma layer 1 (WMMA): sH = relu(x @ g1 + gb1) ---
    for (int ntl = 0; ntl < 2; ++ntl) {
        const int nt = wave * 2 + ntl, n0 = nt * 16;
        const int nn = n0 + (lane & 15);
        v16h bf[8];
        preload_b(wt_g1, n0, lane, bf);
        const float bb = gb1[nn];
        for (int mt = 0; mt < MROWS / 16; ++mt) {
            v8f acc = gemm_tile_b(sK, mt * 16, bf, lane);
#pragma unroll
            for (int vv = 0; vv < 8; ++vv) {
                int m = mt * 16 + vv + hi * 8;
                sH[m * LDS_STRIDE + nn] = (_Float16)fmaxf(acc[vv] + bb, 0.f);
            }
        }
    }
    __syncthreads();

    // --- gamma layer 2 (WMMA): logits -> sK ---
    for (int ntl = 0; ntl < 2; ++ntl) {
        const int nt = wave * 2 + ntl, n0 = nt * 16;
        const int nn = n0 + (lane & 15);
        v16h bf[8];
        preload_b(wt_g2, n0, lane, bf);
        const float bb = gb2[nn];
        for (int mt = 0; mt < MROWS / 16; ++mt) {
            v8f acc = gemm_tile_b(sH, mt * 16, bf, lane);
#pragma unroll
            for (int vv = 0; vv < 8; ++vv) {
                int m = mt * 16 + vv + hi * 8;
                sK[m * LDS_STRIDE + nn] = (_Float16)(acc[vv] + bb);
            }
        }
    }
    __syncthreads();

    // --- softmax over neighbor axis (25 rows) + weighted sum of v+pos ---
    for (int qi = 0; qi < QT; ++qi) {
        const int n = tid;
        const int base = qi * 32;
        float mx = -3.4e38f;
        for (int m = 0; m < NB; ++m)
            mx = fmaxf(mx, (float)sK[(base + m) * LDS_STRIDE + n]);
        float s = 0.f;
        for (int m = 0; m < NB; ++m)
            s += __expf((float)sK[(base + m) * LDS_STRIDE + n] - mx);
        float inv = 1.f / s;
        float o = 0.f;
        for (int m = 0; m < NB; ++m)
            o += __expf((float)sK[(base + m) * LDS_STRIDE + n] - mx)
                 * (float)sV[(base + m) * LDS_STRIDE + n];
        out[(size_t)(gq0 + qi) * DIM + n] = o * inv;
    }
}

// ---------------------------------------------------------------------------
// Host orchestration
// ---------------------------------------------------------------------------
extern "C" void kernel_launch(void* const* d_in, const int* in_sizes, int n_in,
                              void* d_out, int out_size, void* d_ws, size_t ws_size,
                              hipStream_t stream) {
    (void)in_sizes; (void)n_in; (void)out_size; (void)ws_size;

    const float* xyz_q     = (const float*)d_in[0];
    const float* lat_rep   = (const float*)d_in[1];
    const float* xyz       = (const float*)d_in[2];
    const float* points    = (const float*)d_in[3];
    const float* grid_feat = (const float*)d_in[4];
    const float* closest   = (const float*)d_in[5];
    const float* w_qs = (const float*)d_in[6];
    const float* w_ks = (const float*)d_in[7];
    const float* w_vs = (const float*)d_in[8];
    const float* w_kc = (const float*)d_in[9];
    const float* w_vc = (const float*)d_in[10];
    const float* w_kg = (const float*)d_in[11];
    const float* w_vg = (const float*)d_in[12];
    const float* delta_w1 = (const float*)d_in[13];
    const float* delta_b1 = (const float*)d_in[14];
    const float* delta_w2 = (const float*)d_in[15];
    const float* delta_b2 = (const float*)d_in[16];
    const float* gamma_w1 = (const float*)d_in[17];
    const float* gamma_b1 = (const float*)d_in[18];
    const float* gamma_w2 = (const float*)d_in[19];
    const float* gamma_b2 = (const float*)d_in[20];
    float* out = (float*)d_out;

    // Workspace carve
    char* ws = (char*)d_ws;
    size_t off = 0;
    auto alloc = [&](size_t bytes) -> void* {
        void* p = ws + off;
        off = (off + bytes + 255) & ~(size_t)255;
        return p;
    };
    int*      knn  = (int*)alloc((size_t)BB * NQ * KNN * sizeof(int));
    _Float16* pk16 = (_Float16*)alloc((size_t)BB * NN * DIM * sizeof(_Float16));
    _Float16* pv16 = (_Float16*)alloc((size_t)BB * NN * DIM * sizeof(_Float16));
    const size_t wbytes = (size_t)DIM * DIM * sizeof(_Float16);
    _Float16* wt_ks = (_Float16*)alloc(wbytes);
    _Float16* wt_vs = (_Float16*)alloc(wbytes);
    _Float16* wt_kc = (_Float16*)alloc(wbytes);
    _Float16* wt_vc = (_Float16*)alloc(wbytes);
    _Float16* wt_d2 = (_Float16*)alloc(wbytes);
    _Float16* wt_g1 = (_Float16*)alloc(wbytes);
    _Float16* wt_g2 = (_Float16*)alloc(wbytes);
    float* qf  = (float*)alloc((size_t)BB * DIM * sizeof(float));
    float* kgf = (float*)alloc((size_t)BB * DIM * sizeof(float));
    float* vgf = (float*)alloc((size_t)BB * DIM * sizeof(float));

    const int wblocks = (DIM * DIM + 255) / 256;
    prep_weight_kernel<<<wblocks, 256, 0, stream>>>(w_ks, wt_ks);
    prep_weight_kernel<<<wblocks, 256, 0, stream>>>(w_vs, wt_vs);
    prep_weight_kernel<<<wblocks, 256, 0, stream>>>(w_kc, wt_kc);
    prep_weight_kernel<<<wblocks, 256, 0, stream>>>(w_vc, wt_vc);
    prep_weight_kernel<<<wblocks, 256, 0, stream>>>(delta_w2, wt_d2);
    prep_weight_kernel<<<wblocks, 256, 0, stream>>>(gamma_w1, wt_g1);
    prep_weight_kernel<<<wblocks, 256, 0, stream>>>(gamma_w2, wt_g2);

    globals_kernel<<<1, 256, 0, stream>>>(lat_rep, w_qs, w_kg, w_vg, qf, kgf, vgf);

    knn_kernel<<<BB * NQ, 256, 0, stream>>>(xyz_q, xyz, knn);

    dual_proj_kernel<<<(BB * NN) / 64, 256, 0, stream>>>(points, wt_ks, wt_vs,
                                                         pk16, pv16);

    fused_attn_kernel<<<(BB * NQ) / QT, 256, 0, stream>>>(
        knn, pk16, pv16, grid_feat, closest, xyz, xyz_q,
        qf, kgf, vgf, wt_kc, wt_vc,
        delta_w1, delta_b1, wt_d2, delta_b2,
        wt_g1, gamma_b1, wt_g2, gamma_b2, out);
}
